// AttentionCausal2_19859928776720
// MI455X (gfx1250) — matrix operand order repaired and checked by reading, not compiled, not run on hardware
//
#include <hip/hip_runtime.h>
#include <math.h>

// Problem constants (reference: B=4, S=4096, D=1024, fp32 in/out)
#define BB 4
#define SSEQ 4096
#define DDIM 1024
#define MTOT ((size_t)BB * SSEQ)      // 16384 tokens
#define INV_SQRT_D 0.03125f           // 1/sqrt(1024)

typedef __attribute__((ext_vector_type(16))) __bf16 v16bf;
typedef __attribute__((ext_vector_type(8)))  __bf16 v8bf;
typedef __attribute__((ext_vector_type(4)))  __bf16 v4bf;
typedef __attribute__((ext_vector_type(8)))  float  v8f;

// ---- CDNA5 async global->LDS copy (ASYNCcnt-tracked, per-wave counter) ----
__device__ __forceinline__ void async_b128(void* lds, const void* gptr) {
  asm volatile("global_load_async_to_lds_b128 %0, %1, off"
               :: "v"((unsigned)(uintptr_t)lds), "v"(gptr) : "memory");
}
#define WAIT_ASYNC(n) asm volatile("s_wait_asynccnt %0" :: "n"(n) : "memory")

// ---------------------------------------------------------------------------
// Kernel 0: fp32 -> bf16 convert (one-shot; ~5us of HBM traffic)
// ---------------------------------------------------------------------------
__global__ __launch_bounds__(256)
void cvt_f32_bf16(const float* __restrict__ src, __bf16* __restrict__ dst, int n) {
  const int i = (blockIdx.x * 256 + threadIdx.x) * 4;
  if (i < n) {
    const float4 f = *(const float4*)(src + i);
    v4bf o;
    o[0] = (__bf16)f.x; o[1] = (__bf16)f.y; o[2] = (__bf16)f.z; o[3] = (__bf16)f.w;
    *(v4bf*)(dst + i) = o;
  }
}

// ---------------------------------------------------------------------------
// Kernel 1: QKV projection  C[m,n] = sum_d X[m,d] * W[n,d]   (bf16 in/out)
// 128x128 block tile, 8 wave32 waves; wave w owns rows [16w,16w+16).
// Async double-buffered 64-wide K staging; B-fragment loads software-pipelined
// over the WMMAs. z==2 (V) is stored TRANSPOSED: Vt[col][token] so the flash
// kernel's P@V B-fragments are contiguous 32B loads.
// ---------------------------------------------------------------------------
__global__ __launch_bounds__(256)
void qkv_gemm(const __bf16* __restrict__ Xb,
              const __bf16* __restrict__ Wb,     // [3][1024][1024]
              __bf16* __restrict__ Qb,
              __bf16* __restrict__ Kb,
              __bf16* __restrict__ Vt)           // [1024][16384] transposed
{
  __shared__ __bf16 lds_a[2][128][64];   // 16 KB x2  X chunk
  __shared__ __bf16 lds_b[2][128][64];   // 16 KB x2  W chunk

  const __bf16* Wsel = Wb + (size_t)blockIdx.z * DDIM * DDIM;

  const int m0   = blockIdx.x * 128;
  const int n0   = blockIdx.y * 128;
  const int tid  = threadIdx.x;
  const int wave = tid >> 5;        // wave32
  const int lane = tid & 31;
  const int hl   = lane >> 4;
  const int ln   = lane & 15;

  const int srow  = tid >> 1;            // staging row 0..127
  const int scol  = (tid & 1) * 32;      // staging col segment
  const __bf16* ga0 = Xb   + (size_t)(m0 + srow) * DDIM + scol;
  const __bf16* gb0 = Wsel + (size_t)(n0 + srow) * DDIM + scol;

  v8f acc[8] = {};

#define QKV_ISSUE(i)                                                      \
  {                                                                       \
    const int bf_ = (i) & 1;                                              \
    const __bf16* ga = ga0 + (i) * 64;                                    \
    const __bf16* gb = gb0 + (i) * 64;                                    \
    _Pragma("unroll")                                                     \
    for (int u = 0; u < 4; ++u) {                                         \
      async_b128(&lds_a[bf_][srow][scol + 8 * u], ga + 8 * u);            \
      async_b128(&lds_b[bf_][srow][scol + 8 * u], gb + 8 * u);            \
    }                                                                     \
  }

  QKV_ISSUE(0);
  const int NCH = DDIM / 64;             // 16 stages
  for (int i = 0; i < NCH; ++i) {
    if (i + 1 < NCH) { QKV_ISSUE(i + 1); WAIT_ASYNC(8); }
    else             { WAIT_ASYNC(0); }
    __syncthreads();                     // stage i data visible block-wide

    const int buf  = i & 1;
    const int mrow = (wave << 4) + ln;

    auto load_b = [&](int t, int ks) {
      v16bf f;
      const int ncol = (t << 4) + ln;
      const int kb   = ks + (hl << 4);
#pragma unroll
      for (int v = 0; v < 8; ++v) {
        f[2 * v]     = lds_b[buf][ncol][kb + 2 * v];
        f[2 * v + 1] = lds_b[buf][ncol][kb + 2 * v + 1];
      }
      return f;
    };

#pragma unroll
    for (int ks = 0; ks < 64; ks += 32) {
      v16bf afrag;
#pragma unroll
      for (int v = 0; v < 8; ++v) {      // A: 16x32, contiguous K per lane half
        const int kk = ks + (hl << 4) + (v << 1);
        afrag[2 * v]     = lds_a[buf][mrow][kk];
        afrag[2 * v + 1] = lds_a[buf][mrow][kk + 1];
      }
      v16bf bcur = load_b(0, ks);        // pipeline B-frag over WMMA
#pragma unroll
      for (int t = 0; t < 8; ++t) {
        v16bf bnext;
        if (t < 7) bnext = load_b(t + 1, ks);
        acc[t] = __builtin_amdgcn_wmma_f32_16x16x32_bf16(
                   false, afrag, false, bcur, (short)0, acc[t], false, false);
        bcur = bnext;
      }
    }
    __syncthreads();                     // readers done before buf reuse
  }
#undef QKV_ISSUE

  // write-out; C/D layout: lane l, vgpr v -> M = v + 8*(l>>4), N = l&15
  const int mbase = m0 + (wave << 4);
  if (blockIdx.z == 2) {
    // V transposed: Vt[col][token]; each lane's 8 elems are consecutive tokens
#pragma unroll
    for (int t = 0; t < 8; ++t) {
      const int ncol = n0 + (t << 4) + ln;
      v8bf pk;
#pragma unroll
      for (int v = 0; v < 8; ++v) pk[v] = (__bf16)acc[t][v];
      *(v8bf*)(Vt + (size_t)ncol * MTOT + mbase + (hl << 3)) = pk;
    }
  } else {
    __bf16* Out = (blockIdx.z == 0) ? Qb : Kb;
#pragma unroll
    for (int t = 0; t < 8; ++t) {
      const int ncol = n0 + (t << 4) + ln;
#pragma unroll
      for (int v = 0; v < 8; ++v)
        Out[(size_t)(mbase + v + (hl << 3)) * DDIM + ncol] = (__bf16)acc[t][v];
    }
  }
}

// ---------------------------------------------------------------------------
// Kernel 2: causal flash attention, head_dim = 1024.
// One block per (16-query tile, batch); 8 wave32 waves.
//   scores: wave w -> keys [16w,16w+16); K staged in WAVE-PRIVATE LDS double
//           buffers synced only by the wave's own ASYNCcnt (no barriers);
//           dual accumulators break the WMMA dependency chain.
//   P@V   : wave w -> cols [128w,128w+128); V read from transposed Vt as
//           contiguous 32B fragments (L2-resident), pipelined over WMMAs.
// ---------------------------------------------------------------------------
__global__ __launch_bounds__(256)
void flash_attn(const __bf16* __restrict__ Qb,
                const __bf16* __restrict__ Kb,
                const __bf16* __restrict__ Vt,   // [1024][16384]
                float* __restrict__ Out)
{
  __shared__ __bf16 lds_q[16][DDIM];          // 32 KB query tile (shared)
  __shared__ __bf16 lds_k[8][2][16][32];      // 16 KB wave-private K buffers
  __shared__ float  lds_s[16][128];           //  8 KB scores
  __shared__ __bf16 lds_p[16][128];           //  4 KB softmaxed P
  __shared__ float  lds_red[16][16];          //  1 KB partial reductions
  __shared__ float  lds_m[16], lds_l[16], lds_alpha[16];

  const int qi0 = blockIdx.x * 16;
  const int b   = blockIdx.y;
  const size_t base  = (size_t)b * SSEQ * DDIM;
  const size_t btok  = (size_t)b * SSEQ;      // token offset into Vt rows

  const int tid  = threadIdx.x;
  const int wave = tid >> 5;
  const int lane = tid & 31;
  const int hl   = lane >> 4;
  const int ln   = lane & 15;

  // async-load Q tile (16 x 1024 bf16): 128B per thread, one-time barrier
  {
    const int row  = tid >> 4;
    const int col0 = (tid & 15) * 64;
    const __bf16* src = Qb + base + (size_t)(qi0 + row) * DDIM + col0;
#pragma unroll
    for (int u = 0; u < 8; ++u)
      async_b128(&lds_q[row][col0 + 8 * u], src + 8 * u);
  }
  if (tid < 16) { lds_m[tid] = -__builtin_inff(); lds_l[tid] = 0.0f; }
  WAIT_ASYNC(0);
  __syncthreads();

  const int krow = lane >> 1;                 // wave-private staging: row 0..15
  const int kcol = (lane & 1) * 16;           // col segment 0 / 16
  const int srrow = tid >> 4, srseg = tid & 15;  // softmax: row, 8-col segment

  v8f acc[8] = {};
  const int ntiles = (qi0 + 15) / 128 + 1;

  for (int jt = 0; jt < ntiles; ++jt) {
    const int j0 = jt * 128;
    const __bf16* gk0 =
        Kb + base + (size_t)(j0 + (wave << 4) + krow) * DDIM + kcol;

#define K_ISSUE(i)                                                        \
    {                                                                     \
      const int bf_ = (i) & 1;                                            \
      const __bf16* gk = gk0 + (i) * 32;                                  \
      async_b128(&lds_k[wave][bf_][krow][kcol],     gk);                  \
      async_b128(&lds_k[wave][bf_][krow][kcol + 8], gk + 8);              \
    }

    // -------- scores: S = Q K^T over 1024 dims; barrier-free per wave -----
    v8f sacc0 = {}, sacc1 = {};
    K_ISSUE(0);
    for (int i = 0; i < DDIM / 32; ++i) {
      const int kc = i * 32;
      v16bf afrag, bfrag;
#pragma unroll
      for (int v = 0; v < 8; ++v) {       // A = Q[16 x 32] (stable, pre-wait)
        const int kk = kc + (hl << 4) + (v << 1);
        afrag[2 * v]     = lds_q[ln][kk];
        afrag[2 * v + 1] = lds_q[ln][kk + 1];
      }
      if (i + 1 < DDIM / 32) { K_ISSUE(i + 1); WAIT_ASYNC(2); }
      else                   { WAIT_ASYNC(0); }

      const int buf = i & 1;
      const int kb  = hl << 4;            // B = K^T[32 x 16], wave-private rows
#pragma unroll
      for (int v = 0; v < 8; ++v) {
        bfrag[2 * v]     = lds_k[wave][buf][ln][kb + 2 * v];
        bfrag[2 * v + 1] = lds_k[wave][buf][ln][kb + 2 * v + 1];
      }
      if (i & 1)
        sacc1 = __builtin_amdgcn_wmma_f32_16x16x32_bf16(
                  false, afrag, false, bfrag, (short)0, sacc1, false, false);
      else
        sacc0 = __builtin_amdgcn_wmma_f32_16x16x32_bf16(
                  false, afrag, false, bfrag, (short)0, sacc0, false, false);
    }
#undef K_ISSUE

    // scatter scores (scale + causal mask); each wave writes its own 16 cols
    {
      const int keyl = (wave << 4) + ln;
      const int keyg = j0 + keyl;
#pragma unroll
      for (int v = 0; v < 8; ++v) {
        const int row = v + (hl << 3);
        float sv = (sacc0[v] + sacc1[v]) * INV_SQRT_D;
        if (keyg > qi0 + row) sv = -__builtin_inff();
        lds_s[row][keyl] = sv;
      }
    }
    __syncthreads();

    // ---------------- parallel online softmax ----------------------------
    {   // pass 1: partial max (16 rows x 16 segments of 8)
      float mx = -__builtin_inff();
      const int c0 = srseg * 8;
#pragma unroll
      for (int j = 0; j < 8; ++j) mx = fmaxf(mx, lds_s[srrow][c0 + j]);
      lds_red[srrow][srseg] = mx;
    }
    __syncthreads();
    if (tid < 16) {   // pass 2: final max + alpha
      const float m_prev = lds_m[tid];
      float mx = m_prev;
#pragma unroll
      for (int j = 0; j < 16; ++j) mx = fmaxf(mx, lds_red[tid][j]);
      lds_alpha[tid] = __expf(m_prev - mx);
      lds_m[tid] = mx;
    }
    __syncthreads();
    {   // pass 3: exp + partial sums + P (bf16)
      const float m = lds_m[srrow];
      const int c0 = srseg * 8;
      float ps = 0.0f;
#pragma unroll
      for (int j = 0; j < 8; ++j) {
        const float p = __expf(lds_s[srrow][c0 + j] - m);
        lds_p[srrow][c0 + j] = (__bf16)p;
        ps += p;
      }
      lds_red[srrow][srseg] = ps;
    }
    __syncthreads();
    if (tid < 16) {   // pass 4: running denominator
      float ls = lds_l[tid] * lds_alpha[tid];
#pragma unroll
      for (int j = 0; j < 16; ++j) ls += lds_red[tid][j];
      lds_l[tid] = ls;
    }
    __syncthreads();

    // rescale O accumulators
    float arow[8];
#pragma unroll
    for (int v = 0; v < 8; ++v) arow[v] = lds_alpha[v + (hl << 3)];
#pragma unroll
    for (int t = 0; t < 8; ++t)
#pragma unroll
      for (int v = 0; v < 8; ++v) acc[t][v] *= arow[v];

    // -------- O += P @ V: contiguous 32B fragments from transposed Vt ------
    for (int kk = 0; kk < 128; kk += 32) {
      v16bf afrag;
#pragma unroll
      for (int v = 0; v < 8; ++v) {       // A = P[16 x 32]
        const int kq = kk + (hl << 4) + (v << 1);
        afrag[2 * v]     = lds_p[ln][kq];
        afrag[2 * v + 1] = lds_p[ln][kq + 1];
      }
      const size_t koff = btok + (size_t)(j0 + kk + (hl << 4));
      v16bf bcur =
          *(const v16bf*)(Vt + (size_t)((wave << 7) + ln) * MTOT + koff);
#pragma unroll
      for (int t = 0; t < 8; ++t) {
        v16bf bnext;
        if (t < 7)
          bnext = *(const v16bf*)(Vt +
                    (size_t)((wave << 7) + ((t + 1) << 4) + ln) * MTOT + koff);
        acc[t] = __builtin_amdgcn_wmma_f32_16x16x32_bf16(
                   false, afrag, false, bcur, (short)0, acc[t], false, false);
        bcur = bnext;
      }
    }
  }

  // ---------------- finalize: O / l, fp32 store --------------------------
  float lrow[8];
#pragma unroll
  for (int v = 0; v < 8; ++v) lrow[v] = 1.0f / lds_l[v + (hl << 3)];
#pragma unroll
  for (int t = 0; t < 8; ++t) {
    const int col = (wave << 7) + (t << 4) + ln;
#pragma unroll
    for (int v = 0; v < 8; ++v)
      Out[base + (size_t)(qi0 + v + (hl << 3)) * DDIM + col] = acc[t][v] * lrow[v];
  }
}

// ---------------------------------------------------------------------------
extern "C" void kernel_launch(void* const* d_in, const int* in_sizes, int n_in,
                              void* d_out, int out_size, void* d_ws, size_t ws_size,
                              hipStream_t stream) {
  const float* x  = (const float*)d_in[0];   // [4,4096,1024]
  const float* Wq = (const float*)d_in[1];   // [1024,1024]
  const float* Wk = (const float*)d_in[2];
  const float* Wv = (const float*)d_in[3];

  const size_t nX = (size_t)BB * SSEQ * DDIM;   // 16,777,216
  const size_t nW = (size_t)DDIM * DDIM;        //  1,048,576
  __bf16* Xb = (__bf16*)d_ws;                   // bf16 X            (32 MB)
  __bf16* Wb = Xb + nX;                         // bf16 Wq|Wk|Wv     ( 6 MB)
  __bf16* Qb = Wb + 3 * nW;                     // bf16 Q            (32 MB)
  __bf16* Kb = Qb + nX;                         // bf16 K            (32 MB)
  __bf16* Vt = Kb + nX;                         // bf16 V transposed (32 MB)

  cvt_f32_bf16<<<dim3(nX / 1024), 256, 0, stream>>>(x, Xb, (int)nX);
  cvt_f32_bf16<<<dim3(nW / 1024), 256, 0, stream>>>(Wq, Wb,          (int)nW);
  cvt_f32_bf16<<<dim3(nW / 1024), 256, 0, stream>>>(Wk, Wb + nW,     (int)nW);
  cvt_f32_bf16<<<dim3(nW / 1024), 256, 0, stream>>>(Wv, Wb + 2 * nW, (int)nW);

  dim3 g1((BB * SSEQ) / 128, DDIM / 128, 3);    // 128 x 8 x 3
  qkv_gemm<<<g1, 256, 0, stream>>>(Xb, Wb, Qb, Kb, Vt);

  dim3 g2(SSEQ / 16, BB);                       // 256 x 4
  flash_attn<<<g2, 256, 0, stream>>>(Qb, Kb, Vt, (float*)d_out);
}
